// MultiHeadAttention_22050362098253
// MI455X (gfx1250) — compile-verified, matrix operand order checked by reading, not run on hardware
//
#include <hip/hip_runtime.h>
#include <hip/hip_bf16.h>

// ---- problem constants (match reference) ----
#define B_  2
#define S_  2048
#define D_  768
#define H_  12
#define HD_ 64
static constexpr float SCALE = 0.125f; // 64^-0.5

typedef __attribute__((ext_vector_type(16))) _Float16 v16h;
typedef __attribute__((ext_vector_type(8)))  _Float16 v8h;
typedef __attribute__((ext_vector_type(4)))  _Float16 v4h;
typedef __attribute__((ext_vector_type(8)))  float    v8f;
typedef __attribute__((ext_vector_type(4)))  float    v4f;
typedef __attribute__((ext_vector_type(4)))  int      v4i;

union U16h { v16h v; v8h h[2]; };

__device__ __forceinline__ v8f wmma_f16(v16h a, v16h b, v8f c) {
  return __builtin_amdgcn_wmma_f32_16x16x32_f16(false, a, false, b, (short)0, c,
                                                false, false);
}

// ---- CDNA5 async global->LDS copy (ASYNCcnt-tracked), with fallback ----
#if __has_builtin(__builtin_amdgcn_global_load_async_to_lds_b128)
#define HAVE_ASYNC_LDS 1
#endif

__device__ __forceinline__ void cp16B(_Float16* lds, const _Float16* g) {
#ifdef HAVE_ASYNC_LDS
  __builtin_amdgcn_global_load_async_to_lds_b128(
      (__attribute__((address_space(1))) v4i*)g,
      (__attribute__((address_space(3))) v4i*)lds, 0, 0);
#else
  *(v8h*)lds = *(const v8h*)g;
#endif
}

__device__ __forceinline__ void async_wait0() {
#if __has_builtin(__builtin_amdgcn_s_wait_asynccnt)
  __builtin_amdgcn_s_wait_asynccnt(0);
#elif defined(HAVE_ASYNC_LDS)
  asm volatile("s_wait_asynccnt 0x0" ::: "memory");
#endif
}

// ---------------------------------------------------------------------------
// f32 -> f16 conversion kernels (one-time, bandwidth-trivial)
// ---------------------------------------------------------------------------
__global__ void k_cvt(const float* __restrict__ in, _Float16* __restrict__ out,
                      int n4)
{
  int i = blockIdx.x * blockDim.x + threadIdx.x;
  if (i >= n4) return;
  v4f x = *(const v4f*)(in + (size_t)i * 4);
  v4h y;
  y.x = (_Float16)x.x; y.y = (_Float16)x.y;
  y.z = (_Float16)x.z; y.w = (_Float16)x.w;
  *(v4h*)(out + (size_t)i * 4) = y;
}

// out[n*K + k] = (f16) in[k*N + n]   (store weights transposed, N x K)
__global__ void k_cvt_tr(const float* __restrict__ in, _Float16* __restrict__ out,
                         int K, int N)
{
  int i = blockIdx.x * blockDim.x + threadIdx.x;
  if (i >= K * N) return;
  int n = i / K, k = i % K;
  out[i] = (_Float16)in[(size_t)k * N + n];
}

// ---------------------------------------------------------------------------
// GEMM core: C(64x64 block) = A(MxK f16, row-major) * BT(NxK f16)^T
// 4 waves, each 16(M) x 64(N); K-step 32; double-buffered async LDS staging.
// ---------------------------------------------------------------------------
#define GEMM_BODY(K_DIM)                                                       \
  __shared__ alignas(16) _Float16 As[2][64 * 32];                              \
  __shared__ alignas(16) _Float16 Bs[2][64 * 32];                              \
  const int t      = threadIdx.x;                                              \
  const int lane   = t & 31;                                                   \
  const int wave   = t >> 5;                                                   \
  const int m0     = blockIdx.x * 64;                                          \
  const int n0     = blockIdx.y * 64;                                          \
  const int row_in = lane & 15;                                                \
  const int half   = lane >> 4;                                                \
  const int kb     = half << 3;                                                \
  const int e0     = t * 16;                                                   \
  const int sr     = e0 >> 5;       /* staging row (0..63)      */             \
  const int sc     = e0 & 31;       /* staging col (0 or 16)    */             \
  v8f acc[4] = {};                                                             \
  const int nk = (K_DIM) / 32;                                                 \
  /* prologue stage of tile 0 */                                               \
  {                                                                            \
    const _Float16* ga = &A [(size_t)(m0 + sr) * (K_DIM) + sc];                \
    const _Float16* gb = &BT[(size_t)(n0 + sr) * (K_DIM) + sc];                \
    cp16B(&As[0][e0], ga);  cp16B(&As[0][e0 + 8], ga + 8);                     \
    cp16B(&Bs[0][e0], gb);  cp16B(&Bs[0][e0 + 8], gb + 8);                     \
  }                                                                            \
  for (int kt = 0; kt < nk; ++kt) {                                            \
    const int buf = kt & 1;                                                    \
    async_wait0();                                                             \
    __syncthreads();                                                           \
    if (kt + 1 < nk) {                                                         \
      const int k0 = (kt + 1) * 32;                                            \
      const _Float16* ga = &A [(size_t)(m0 + sr) * (K_DIM) + k0 + sc];         \
      const _Float16* gb = &BT[(size_t)(n0 + sr) * (K_DIM) + k0 + sc];         \
      cp16B(&As[buf ^ 1][e0], ga);  cp16B(&As[buf ^ 1][e0 + 8], ga + 8);       \
      cp16B(&Bs[buf ^ 1][e0], gb);  cp16B(&Bs[buf ^ 1][e0 + 8], gb + 8);       \
    }                                                                          \
    U16h af;                                                                   \
    const _Float16* arow = &As[buf][(wave * 16 + row_in) * 32];                \
    af.h[0] = *(const v8h*)(arow + kb);                                        \
    af.h[1] = *(const v8h*)(arow + 16 + kb);                                   \
    _Pragma("unroll")                                                          \
    for (int nt = 0; nt < 4; ++nt) {                                           \
      U16h bf;                                                                 \
      const _Float16* brow = &Bs[buf][(nt * 16 + row_in) * 32];                \
      bf.h[0] = *(const v8h*)(brow + kb);                                      \
      bf.h[1] = *(const v8h*)(brow + 16 + kb);                                 \
      acc[nt] = wmma_f16(af.v, bf.v, acc[nt]);                                 \
    }                                                                          \
  }

// QKV projection: A = xh (4096x768), BT = wqkvT (2304x768); scatter f16 Q/K/V.
__global__ __launch_bounds__(128)
void k_gemm_qkv(const _Float16* __restrict__ A, const _Float16* __restrict__ BT,
                _Float16* __restrict__ q16, _Float16* __restrict__ k16,
                _Float16* __restrict__ v16)
{
  GEMM_BODY(D_)
  for (int nt = 0; nt < 4; ++nt) {
    int j     = n0 + nt * 16 + row_in;   // column in [0, 2304)
    int which = j / D_;
    int jj    = j % D_;
    int h     = jj >> 6;
    int hd    = jj & 63;
    _Float16* dst = (which == 0) ? q16 : (which == 1) ? k16 : v16;
#pragma unroll
    for (int i = 0; i < 8; ++i) {
      int row = m0 + wave * 16 + i + half * 8;   // b*S + s
      int b   = row >> 11;
      int s   = row & (S_ - 1);
      dst[((size_t)((b * H_ + h) * S_ + s)) * HD_ + hd] = (_Float16)acc[nt][i];
    }
  }
}

// Output projection: A = attn16 (4096x768), BT = wprojT (768x768); f32 out.
__global__ __launch_bounds__(128)
void k_gemm_proj(const _Float16* __restrict__ A, const _Float16* __restrict__ BT,
                 float* __restrict__ out)
{
  GEMM_BODY(D_)
  for (int nt = 0; nt < 4; ++nt) {
    int col = n0 + nt * 16 + row_in;
#pragma unroll
    for (int i = 0; i < 8; ++i) {
      int row = m0 + wave * 16 + i + half * 8;
      out[(size_t)row * D_ + col] = acc[nt][i];
    }
  }
}

// ---------------------------------------------------------------------------
// RoPE in place on Q and K, layout (B*H, S, HD), half-split rotary.
// ---------------------------------------------------------------------------
__global__ void k_rope(_Float16* __restrict__ q16, _Float16* __restrict__ k16)
{
  int idx = blockIdx.x * blockDim.x + threadIdx.x;
  const int total = B_ * H_ * S_ * (HD_ / 2);
  if (idx >= total) return;
  int d  = idx & 31;
  int s  = (idx >> 5) & (S_ - 1);
  int bh = idx >> 16;
  size_t base = ((size_t)bh * S_ + s) * HD_;

  float inv_freq = __expf(-logf(10000.0f) * (float)d * (1.0f / 32.0f));
  float theta = (float)s * inv_freq;
  float sn, cs;
  __sincosf(theta, &sn, &cs);

  {
    float x1 = (float)q16[base + d];
    float x2 = (float)q16[base + d + 32];
    q16[base + d]      = (_Float16)( x1 * cs + x2 * sn);
    q16[base + d + 32] = (_Float16)(-x1 * sn + x2 * cs);
  }
  {
    float x1 = (float)k16[base + d];
    float x2 = (float)k16[base + d + 32];
    k16[base + d]      = (_Float16)( x1 * cs + x2 * sn);
    k16[base + d + 32] = (_Float16)(-x1 * sn + x2 * cs);
  }
}

// ---------------------------------------------------------------------------
// Causal flash attention. Block = 4 waves = 64 query rows of one (b,h).
// K,V staged row-major via double-buffered async LDS copies.
// ---------------------------------------------------------------------------
__global__ __launch_bounds__(128)
void k_attn(const _Float16* __restrict__ q16, const _Float16* __restrict__ k16,
            const _Float16* __restrict__ v16, _Float16* __restrict__ attn16)
{
  __shared__ alignas(16) _Float16 Kb[2][32 * 64];   // [key][hd]
  __shared__ alignas(16) _Float16 Vb[2][32 * 64];   // [key][hd]
  __shared__ alignas(16) _Float16 Pt[4][16 * 32];   // per-wave P tile

  const int t      = threadIdx.x;
  const int lane   = t & 31;
  const int wave   = t >> 5;
  const int nqb    = S_ / 64;
  const int bh     = blockIdx.x / nqb;
  const int qblk   = blockIdx.x % nqb;
  const int q0b    = qblk * 64;
  const int q0     = q0b + wave * 16;
  const int row_in = lane & 15;
  const int half   = lane >> 4;
  const int kb     = half << 3;
  const size_t base = (size_t)bh * S_ * HD_;
  const int e0  = t * 16;
  const int key = e0 >> 6, hd0 = e0 & 63;           // staging coords

  // Q fragments (hd 0..31 and 32..63), resident for the whole KV loop
  U16h a0, a1;
  {
    const _Float16* qrow = &q16[base + (size_t)(q0 + row_in) * HD_];
    a0.h[0] = *(const v8h*)(qrow + kb);
    a0.h[1] = *(const v8h*)(qrow + 16 + kb);
    a1.h[0] = *(const v8h*)(qrow + 32 + kb);
    a1.h[1] = *(const v8h*)(qrow + 48 + kb);
  }

  float m_i[8], l_i[8];
  v8f o[4] = {};
  for (int i = 0; i < 8; ++i) { m_i[i] = -1e30f; l_i[i] = 0.0f; }

  const int kend = q0b + 64;
  // prologue stage of tile 0
  {
    const _Float16* ks = &k16[base + (size_t)key * HD_ + hd0];
    const _Float16* vs = &v16[base + (size_t)key * HD_ + hd0];
    cp16B(&Kb[0][e0], ks);  cp16B(&Kb[0][e0 + 8], ks + 8);
    cp16B(&Vb[0][e0], vs);  cp16B(&Vb[0][e0 + 8], vs + 8);
  }

  for (int kv0 = 0; kv0 < kend; kv0 += 32) {
    const int buf = (kv0 >> 5) & 1;
    async_wait0();
    __syncthreads();
    if (kv0 + 32 < kend) {
      const _Float16* ks = &k16[base + (size_t)(kv0 + 32 + key) * HD_ + hd0];
      const _Float16* vs = &v16[base + (size_t)(kv0 + 32 + key) * HD_ + hd0];
      cp16B(&Kb[buf ^ 1][e0], ks);  cp16B(&Kb[buf ^ 1][e0 + 8], ks + 8);
      cp16B(&Vb[buf ^ 1][e0], vs);  cp16B(&Vb[buf ^ 1][e0 + 8], vs + 8);
    }

    if (kv0 <= q0 + 15) {                           // tile in causal region
      // ---- S = Q @ K^T  (16q x 32k), K-depth 64 ----
      v8f sc[2];
#pragma unroll
      for (int j = 0; j < 2; ++j) {
        U16h b0, b1;
        const _Float16* krow = &Kb[buf][(j * 16 + row_in) * 64];
        b0.h[0] = *(const v8h*)(krow + kb);
        b0.h[1] = *(const v8h*)(krow + 16 + kb);
        b1.h[0] = *(const v8h*)(krow + 32 + kb);
        b1.h[1] = *(const v8h*)(krow + 48 + kb);
        v8f s = {};
        s = wmma_f16(a0.v, b0.v, s);
        s = wmma_f16(a1.v, b1.v, s);
        sc[j] = s;
      }

      // ---- online softmax (row = reg i + 8*half, col = lane&15) ----
      float p[2][8];
#pragma unroll
      for (int i = 0; i < 8; ++i) {
        int qrow = q0 + i + half * 8;
        float mx = -1e30f;
#pragma unroll
        for (int j = 0; j < 2; ++j) {
          int kidx = kv0 + j * 16 + row_in;
          float v = sc[j][i] * SCALE;
          if (kidx > qrow) v = -1e30f;              // causal mask
          p[j][i] = v;
          mx = fmaxf(mx, v);
        }
        for (int m = 8; m >= 1; m >>= 1) mx = fmaxf(mx, __shfl_xor(mx, m, 32));
        float m_new = fmaxf(m_i[i], mx);
        float alpha = __expf(m_i[i] - m_new);
        float rs = 0.0f;
#pragma unroll
        for (int j = 0; j < 2; ++j) {
          float e = __expf(p[j][i] - m_new);
          p[j][i] = e;
          rs += e;
        }
        for (int m = 8; m >= 1; m >>= 1) rs += __shfl_xor(rs, m, 32);
        l_i[i] = l_i[i] * alpha + rs;
        m_i[i] = m_new;
#pragma unroll
        for (int tt = 0; tt < 4; ++tt) o[tt][i] *= alpha;
      }

      // ---- re-layout P: C-layout -> A-fragment via per-wave LDS ----
      _Float16* P = Pt[wave];
#pragma unroll
      for (int i = 0; i < 8; ++i) {
        int r = i + half * 8;
        P[r * 32 + row_in]      = (_Float16)p[0][i];
        P[r * 32 + 16 + row_in] = (_Float16)p[1][i];
      }
      asm volatile("s_wait_dscnt 0" ::: "memory");  // same-wave LDS RAW fence

      U16h pf;
      const _Float16* prow = &P[row_in * 32];
      pf.h[0] = *(const v8h*)(prow + kb);
      pf.h[1] = *(const v8h*)(prow + 16 + kb);

      // ---- O += P @ V : B-frags gathered from row-major V tile ----
#pragma unroll
      for (int tt = 0; tt < 4; ++tt) {
        U16h vf;
        const _Float16* vcol = &Vb[buf][tt * 16 + row_in];
#pragma unroll
        for (int e = 0; e < 8; ++e) vf.v[e]     = vcol[(kb + e) * 64];
#pragma unroll
        for (int e = 0; e < 8; ++e) vf.v[e + 8] = vcol[(16 + kb + e) * 64];
        o[tt] = wmma_f16(pf.v, vf.v, o[tt]);
      }
    }
  }

  // ---- epilogue: normalize, write (B,S,H,HD) f16 ----
  const int b = bh / H_, h = bh % H_;
#pragma unroll
  for (int i = 0; i < 8; ++i) {
    float inv_l = 1.0f / l_i[i];
    int s = q0 + i + half * 8;
    size_t orow = ((size_t)(b * S_ + s) * H_ + h) * HD_;
#pragma unroll
    for (int tt = 0; tt < 4; ++tt)
      attn16[orow + tt * 16 + row_in] = (_Float16)(o[tt][i] * inv_l);
  }
}

// ---------------------------------------------------------------------------
extern "C" void kernel_launch(void* const* d_in, const int* in_sizes, int n_in,
                              void* d_out, int out_size, void* d_ws, size_t ws_size,
                              hipStream_t stream)
{
  (void)in_sizes; (void)n_in; (void)out_size; (void)ws_size;
  const float* x      = (const float*)d_in[0];   // (B,S,D) f32
  const float* w_qkv  = (const float*)d_in[1];   // (D,3D) f32
  const float* w_proj = (const float*)d_in[2];   // (D,D) f32
  float* out = (float*)d_out;

  _Float16* ws = (_Float16*)d_ws;
  const size_t per = (size_t)B_ * H_ * S_ * HD_; // 3,145,728 halfs
  _Float16* q16    = ws;
  _Float16* k16    = ws + per;
  _Float16* v16    = ws + 2 * per;
  _Float16* attn16 = ws + 3 * per;
  _Float16* xh     = ws + 4 * per;                       // (4096 x 768)
  _Float16* wqkvT  = xh + (size_t)B_ * S_ * D_;          // (2304 x 768)
  _Float16* wprojT = wqkvT + (size_t)(3 * D_) * D_;      // (768 x 768)

  // one-time f16 conversion (+ weight transpose)
  {
    int n4 = (B_ * S_ * D_) / 4;
    k_cvt<<<(n4 + 255) / 256, 256, 0, stream>>>(x, xh, n4);
    int nq = D_ * 3 * D_;
    k_cvt_tr<<<(nq + 255) / 256, 256, 0, stream>>>(w_qkv, wqkvT, D_, 3 * D_);
    int np = D_ * D_;
    k_cvt_tr<<<(np + 255) / 256, 256, 0, stream>>>(w_proj, wprojT, D_, D_);
  }

  dim3 g1((B_ * S_) / 64, (3 * D_) / 64);        // 64 x 36
  k_gemm_qkv<<<g1, 128, 0, stream>>>(xh, wqkvT, q16, k16, v16);

  int rope_total = B_ * H_ * S_ * (HD_ / 2);
  k_rope<<<(rope_total + 255) / 256, 256, 0, stream>>>(q16, k16);

  k_attn<<<(B_ * H_) * (S_ / 64), 128, 0, stream>>>(q16, k16, v16, attn16);

  dim3 g2((B_ * S_) / 64, D_ / 64);              // 64 x 12
  k_gemm_proj<<<g2, 128, 0, stream>>>(attn16, wprojT, out);
}
